// ResponseDecoder_20701742367085
// MI455X (gfx1250) — matrix operand order repaired and checked by reading, not compiled, not run on hardware
//
#include <hip/hip_runtime.h>
#include <math.h>

// ---------------------------------------------------------------------------
// ResponseDecoder step for MI455X (gfx1250).
// Heavy GEMMs use V_WMMA_F32_16X16X32_BF16 (f32 -> bf16 on the fly, f32 acc),
// double-buffered LDS staging, unguarded vectorized fast path for aligned dims.
// Shapes: TZ=TU=512, B=64, H=512, E=256, V=3000, D=5.
// ---------------------------------------------------------------------------

#define TZ 512
#define TU 512
#define Bz 64
#define Hh 512
#define Ee 256
#define Vv 3000
#define Dd 5
#define VT (Vv + TZ)          // 3512

typedef __attribute__((ext_vector_type(16))) __bf16 v16bf;
typedef __attribute__((ext_vector_type(8)))  float  v8f;

union Frag16 { unsigned short u[16]; v16bf v; };

static __device__ __forceinline__ unsigned short f32_to_bf16(float f) {
    union { float f; unsigned u; } x; x.f = f;
    unsigned r = x.u + 0x7FFFu + ((x.u >> 16) & 1u);   // round-to-nearest-even
    return (unsigned short)(r >> 16);
}

static __device__ __forceinline__ unsigned long long pack4bf16(float4 v) {
    unsigned u0 = (unsigned)f32_to_bf16(v.x) | ((unsigned)f32_to_bf16(v.y) << 16);
    unsigned u1 = (unsigned)f32_to_bf16(v.z) | ((unsigned)f32_to_bf16(v.w) << 16);
    return (unsigned long long)u0 | ((unsigned long long)u1 << 32);
}

// ---------------------------------------------------------------------------
// Stage one 64x32 A tile + 64x32 Bt tile (bf16) into LDS.
// GUARD=false: all dims tile-aligned, vectorized float4 loads (lda/ldb must be
// multiples of 4 floats -> 16B-aligned rows; true for all unguarded callers).
// BT=true: B is (N x K) row-major; BT=false: B is (K x N) row-major.
// ---------------------------------------------------------------------------
template <bool BT, bool GUARD>
static __device__ __forceinline__ void stage_tile(
    unsigned short (&As)[64][40], unsigned short (&Bs)[64][40],
    const float* __restrict__ A, int lda,
    const float* __restrict__ B, int ldb,
    int m0, int n0, int k0, int M, int N, int K, int tid)
{
    // ---- A tile: rows = m, cols = k (contiguous) ----
    {
        int row = tid >> 3;             // 0..15
        int col = (tid & 7) * 4;        // 0,4,...,28
#pragma unroll
        for (int p = 0; p < 4; ++p) {
            int r = p * 16 + row;
            if (GUARD) {
                int gr = m0 + r;
#pragma unroll
                for (int c = 0; c < 4; ++c) {
                    int gk = k0 + col + c;
                    float vv = (gr < M && gk < K) ? A[(long)gr * lda + gk] : 0.0f;
                    As[r][col + c] = f32_to_bf16(vv);
                }
            } else {
                float4 v4 = *(const float4*)(A + (long)(m0 + r) * lda + k0 + col);
                *(unsigned long long*)&As[r][col] = pack4bf16(v4);
            }
        }
    }
    // ---- B tile into Bt layout Bs[n][k] ----
    if (BT) {
        // B rows are n, cols are k (contiguous) -> same pattern as A
        int row = tid >> 3;
        int col = (tid & 7) * 4;
#pragma unroll
        for (int p = 0; p < 4; ++p) {
            int n = p * 16 + row;
            if (GUARD) {
                int gn = n0 + n;
#pragma unroll
                for (int c = 0; c < 4; ++c) {
                    int gk = k0 + col + c;
                    float vv = (gn < N && gk < K) ? B[(long)gn * ldb + gk] : 0.0f;
                    Bs[n][col + c] = f32_to_bf16(vv);
                }
            } else {
                float4 v4 = *(const float4*)(B + (long)(n0 + n) * ldb + k0 + col);
                *(unsigned long long*)&Bs[n][col] = pack4bf16(v4);
            }
        }
    } else {
        // B rows are k, cols are n (contiguous) -> read float4 along n, scatter k
        int kk = tid >> 4;              // 0..7
        int nn = (tid & 15) * 4;        // 0,4,...,60
#pragma unroll
        for (int p = 0; p < 4; ++p) {
            int k = p * 8 + kk;
            if (GUARD) {
                int gk = k0 + k;
#pragma unroll
                for (int c = 0; c < 4; ++c) {
                    int gn = n0 + nn + c;
                    float vv = (gk < K && gn < N) ? B[(long)gk * ldb + gn] : 0.0f;
                    Bs[nn + c][k] = f32_to_bf16(vv);
                }
            } else {
                float4 v4 = *(const float4*)(B + (long)(k0 + k) * ldb + n0 + nn);
                Bs[nn + 0][k] = f32_to_bf16(v4.x);
                Bs[nn + 1][k] = f32_to_bf16(v4.y);
                Bs[nn + 2][k] = f32_to_bf16(v4.z);
                Bs[nn + 3][k] = f32_to_bf16(v4.w);
            }
        }
    }
}

// ---------------------------------------------------------------------------
// bf16-WMMA GEMM, 64x64 C tile per 128-thread workgroup (4 waves), K step 32,
// double-buffered LDS.  MODE 0: C = A@B (+bias[n]).  MODE 1: fused epilogue
//   out[r] += sum_n tanh(C[r,n] + addv[(r%rowMod)*addStride + n])
//                  * mulv[(r%rowMod)*mulStride + n]      (atomicAdd partials)
// ---------------------------------------------------------------------------
template <int MODE, bool BT, bool GUARD>
__global__ __launch_bounds__(128)
void gemm_wmma(const float* __restrict__ A, int lda,
               const float* __restrict__ B, int ldb,
               float* __restrict__ Cout, int ldc,
               const float* __restrict__ bias,
               const float* __restrict__ addv, int addStride,
               const float* __restrict__ mulv, int mulStride,
               float* __restrict__ outRed,
               int M, int N, int K, int rowMod)
{
    const int BK = 32;
    __shared__ unsigned short As[2][64][40];   // padded rows: 80 B (16B aligned)
    __shared__ unsigned short Bs[2][64][40];   // Bt layout: [n][k]

    const int tid  = threadIdx.x;
    const int wave = tid >> 5;
    const int lane = tid & 31;
    const int l16  = lane & 15;
    const int half = lane >> 4;
    const int m0   = blockIdx.y * 64;
    const int n0   = blockIdx.x * 64;

    v8f acc[4];
#pragma unroll
    for (int i = 0; i < 4; ++i)
#pragma unroll
        for (int j = 0; j < 8; ++j) acc[i][j] = 0.0f;

    const int kTiles = (K + BK - 1) / BK;
    stage_tile<BT, GUARD>(As[0], Bs[0], A, lda, B, ldb, m0, n0, 0, M, N, K, tid);

    for (int kt = 0; kt < kTiles; ++kt) {
        __syncthreads();
        const int cur = kt & 1;
        if (kt + 1 < kTiles)
            stage_tile<BT, GUARD>(As[cur ^ 1], Bs[cur ^ 1], A, lda, B, ldb,
                                  m0, n0, (kt + 1) * BK, M, N, K, tid);

        // A fragment (16-bit 16x32 layout): lanes<16 -> K 0-7 & 16-23; lanes>=16 -> 8-15 & 24-31
        Frag16 a;
        {
            int r  = wave * 16 + l16;
            int kb = half * 8;
#pragma unroll
            for (int i = 0; i < 8; ++i) a.u[i]     = As[cur][r][kb + i];
#pragma unroll
            for (int i = 0; i < 8; ++i) a.u[8 + i] = As[cur][r][kb + 16 + i];
        }
        // All 4 B fragments first, then back-to-back WMMAs
        Frag16 bf[4];
#pragma unroll
        for (int nt = 0; nt < 4; ++nt) {
            int n  = nt * 16 + l16;
            int kb = half * 8;
#pragma unroll
            for (int i = 0; i < 8; ++i) bf[nt].u[i]     = Bs[cur][n][kb + i];
#pragma unroll
            for (int i = 0; i < 8; ++i) bf[nt].u[8 + i] = Bs[cur][n][kb + 16 + i];
        }
#pragma unroll
        for (int nt = 0; nt < 4; ++nt)
            acc[nt] = __builtin_amdgcn_wmma_f32_16x16x32_bf16(
                false, a.v, false, bf[nt].v, (short)0, acc[nt], false, false);
    }

    if (MODE == 0) {
#pragma unroll
        for (int nt = 0; nt < 4; ++nt) {
            int gn = n0 + nt * 16 + l16;
            if (gn >= N) continue;
            float bsv = bias ? bias[gn] : 0.0f;
#pragma unroll
            for (int v = 0; v < 8; ++v) {
                int gr = m0 + wave * 16 + half * 8 + v;
                if (gr < M) Cout[(long)gr * ldc + gn] = acc[nt][v] + bsv;
            }
        }
    } else {
        float rsum[8];
#pragma unroll
        for (int v = 0; v < 8; ++v) rsum[v] = 0.0f;
#pragma unroll
        for (int nt = 0; nt < 4; ++nt) {
            int gn = n0 + nt * 16 + l16;
            bool nok = (gn < N);
#pragma unroll
            for (int v = 0; v < 8; ++v) {
                int gr = m0 + wave * 16 + half * 8 + v;
                if (nok && gr < M) {
                    int bb = gr % rowMod;
                    float av = addv ? addv[bb * addStride + gn] : 0.0f;
                    float mv = mulv ? mulv[bb * mulStride + gn] : 1.0f;
                    rsum[v] += tanhf(acc[nt][v] + av) * mv;
                }
            }
        }
        // reduce across the 16 lanes of each half (C layout: same M rows per half)
#pragma unroll
        for (int v = 0; v < 8; ++v) {
            float s = rsum[v];
            s += __shfl_xor(s, 8, 32);
            s += __shfl_xor(s, 4, 32);
            s += __shfl_xor(s, 2, 32);
            s += __shfl_xor(s, 1, 32);
            if (l16 == 0) {
                int gr = m0 + wave * 16 + half * 8 + v;
                if (gr < M) atomicAdd(&outRed[gr], s);
            }
        }
    }
}

// ---------------------------------------------------------------------------
__global__ void k_zero(float* p, int n) {
    int i = blockIdx.x * 256 + threadIdx.x;
    if (i < n) p[i] = 0.0f;
}

__global__ void k_embed(const int* __restrict__ mt, const float* __restrict__ embW,
                        float* __restrict__ out) {
    int b = blockIdx.x, e = threadIdx.x;              // 64 blocks x 256 threads
    out[b * Ee + e] = embW[(long)mt[b] * Ee + e];
}

// softmax over t of e[t*B+b], then ctx[b,h] = sum_t a_t * enc[t,b,h]
__global__ __launch_bounds__(256)
void k_attn_ctx(const float* __restrict__ e, const float* __restrict__ enc,
                float* __restrict__ ctx) {
    int b = blockIdx.x, tid = threadIdx.x;
    __shared__ float a[TZ];
    __shared__ float red[256];
    float mx = -INFINITY;
    for (int t = tid; t < TZ; t += 256) mx = fmaxf(mx, e[t * Bz + b]);
    red[tid] = mx; __syncthreads();
    for (int s = 128; s > 0; s >>= 1) { if (tid < s) red[tid] = fmaxf(red[tid], red[tid + s]); __syncthreads(); }
    mx = red[0]; __syncthreads();
    float sm = 0.0f;
    for (int t = tid; t < TZ; t += 256) { float v = expf(e[t * Bz + b] - mx); a[t] = v; sm += v; }
    red[tid] = sm; __syncthreads();
    for (int s = 128; s > 0; s >>= 1) { if (tid < s) red[tid] += red[tid + s]; __syncthreads(); }
    float inv = 1.0f / red[0]; __syncthreads();
    for (int h = tid; h < Hh; h += 256) {
        float s = 0.0f;
        for (int t = 0; t < TZ; ++t) s += a[t] * enc[(long)t * (Bz * Hh) + b * Hh + h];
        ctx[b * Hh + h] = s * inv;
    }
}

__global__ void k_concat_gru(const float* memb, const float* uctx, const float* zctx,
                             const float* deg, float* out) {
    int b = blockIdx.x;
    for (int i = threadIdx.x; i < Ee + 2 * Hh + Dd; i += 256) {
        float v;
        if (i < Ee)             v = memb[b * Ee + i];
        else if (i < Ee + Hh)   v = uctx[b * Hh + (i - Ee)];
        else if (i < Ee + 2*Hh) v = zctx[b * Hh + (i - Ee - Hh)];
        else                    v = deg[b * Dd + (i - Ee - 2 * Hh)];
        out[b * (Ee + 2 * Hh + Dd) + i] = v;
    }
}

__global__ void k_gru_gate(const float* gi, const float* gh, const float* h0,
                           float* h1, float* oA, float* oB) {
    int b = blockIdx.x;
    for (int j = threadIdx.x; j < Hh; j += 256) {
        float ir = gi[b * 3 * Hh + j], iz = gi[b * 3 * Hh + Hh + j], in = gi[b * 3 * Hh + 2 * Hh + j];
        float hr = gh[b * 3 * Hh + j], hz = gh[b * 3 * Hh + Hh + j], hn = gh[b * 3 * Hh + 2 * Hh + j];
        float r = 1.0f / (1.0f + expf(-(ir + hr)));
        float z = 1.0f / (1.0f + expf(-(iz + hz)));
        float n = tanhf(in + r * hn);
        float h = (1.0f - z) * n + z * h0[b * Hh + j];
        h1[b * Hh + j] = h; oA[b * Hh + j] = h; oB[b * Hh + j] = h;
    }
}

__global__ void k_concat_proj(const float* zctx, const float* uctx, const float* h1,
                              float* out) {
    int b = blockIdx.x;
    for (int i = threadIdx.x; i < 3 * Hh; i += 256) {
        float v;
        if (i < Hh)            v = zctx[b * Hh + i];
        else if (i < 2 * Hh)   v = uctx[b * Hh + (i - Hh)];
        else                   v = h1[b * Hh + (i - 2 * Hh)];
        out[b * 3 * Hh + i] = v;
    }
}

__global__ __launch_bounds__(256)
void k_ecs(const float* __restrict__ cs, float* __restrict__ ecs, float* __restrict__ mbuf) {
    int b = blockIdx.x, tid = threadIdx.x;
    __shared__ float red[256];
    float mx = -INFINITY;
    for (int t = tid; t < TZ; t += 256) mx = fmaxf(mx, cs[t * Bz + b]);
    red[tid] = mx; __syncthreads();
    for (int s = 128; s > 0; s >>= 1) { if (tid < s) red[tid] = fmaxf(red[tid], red[tid + s]); __syncthreads(); }
    mx = red[0];
    if (tid == 0) mbuf[b] = mx;
    for (int t = tid; t < TZ; t += 256) ecs[b * TZ + t] = expf(cs[t * Bz + b] - mx);
}

// z_copy[b,v] = log(sum_t ecs[b,t]*sparse[b,t,v]) + m[b]
// HBM streaming of 460 MB via float4 (VT % 4 == 0, row stride 14048 B 16B-aligned)
__global__ __launch_bounds__(256)
void k_copy(const float* __restrict__ ecs, const float* __restrict__ mbuf,
            const float* __restrict__ sparse, float* __restrict__ zcopy) {
    int b = blockIdx.y;
    int v4 = blockIdx.x * 256 + threadIdx.x;          // float4 column index
    __shared__ float w[TZ];
    for (int t = threadIdx.x; t < TZ; t += 256) w[t] = ecs[b * TZ + t];
    __syncthreads();
    if (v4 >= VT / 4) return;
    const float4* sp = (const float4*)(sparse + (long)b * TZ * VT);
    float4 s = make_float4(0.f, 0.f, 0.f, 0.f);
    for (int t = 0; t < TZ; ++t) {
        float4 x = sp[(long)t * (VT / 4) + v4];
        float wt = w[t];
        s.x += wt * x.x; s.y += wt * x.y; s.z += wt * x.z; s.w += wt * x.w;
    }
    float m = mbuf[b];
    float4 o;
    o.x = logf(s.x) + m; o.y = logf(s.y) + m; o.z = logf(s.z) + m; o.w = logf(s.w) + m;
    ((float4*)(zcopy + (long)b * VT))[v4] = o;
}

__global__ __launch_bounds__(256)
void k_final(const float* __restrict__ gen, const float* __restrict__ zcopy,
             float* __restrict__ proba) {
    int b = blockIdx.x, tid = threadIdx.x;
    __shared__ float red[256];
    const int NL = Vv + VT;  // 6512 logits
    float mx = -INFINITY;
    for (int i = tid; i < NL; i += 256) {
        float v = (i < Vv) ? gen[b * Vv + i] : zcopy[b * VT + (i - Vv)];
        mx = fmaxf(mx, v);
    }
    red[tid] = mx; __syncthreads();
    for (int s = 128; s > 0; s >>= 1) { if (tid < s) red[tid] = fmaxf(red[tid], red[tid + s]); __syncthreads(); }
    mx = red[0]; __syncthreads();
    float sm = 0.0f;
    for (int i = tid; i < NL; i += 256) {
        float v = (i < Vv) ? gen[b * Vv + i] : zcopy[b * VT + (i - Vv)];
        sm += expf(v - mx);
    }
    red[tid] = sm; __syncthreads();
    for (int s = 128; s > 0; s >>= 1) { if (tid < s) red[tid] += red[tid + s]; __syncthreads(); }
    float inv = 1.0f / red[0];
    for (int i = tid; i < VT; i += 256) {
        float c = expf(zcopy[b * VT + i] - mx) * inv;
        float g = (i < Vv) ? expf(gen[b * Vv + i] - mx) * inv : 0.0f;
        proba[b * VT + i] = g + c;
    }
}

// ---------------------------------------------------------------------------
extern "C" void kernel_launch(void* const* d_in, const int* in_sizes, int n_in,
                              void* d_out, int out_size, void* d_ws, size_t ws_size,
                              hipStream_t stream) {
    const float* z_enc  = (const float*)d_in[0];   // (512,64,512)
    const float* u_enc  = (const float*)d_in[1];   // (512,64,512)
    const int*   m_t    = (const int*)  d_in[2];   // (1,64)
    const float* degree = (const float*)d_in[3];   // (64,5)
    const float* h0     = (const float*)d_in[4];   // (1,64,512)
    const float* sparse = (const float*)d_in[5];   // (64,512,3512)
    const float* embW   = (const float*)d_in[6];   // (3000,256)
    const float* Wz     = (const float*)d_in[7];   // (1024,512)
    const float* bz     = (const float*)d_in[8];
    const float* vz     = (const float*)d_in[9];
    const float* Wu     = (const float*)d_in[10];
    const float* bu     = (const float*)d_in[11];
    const float* vu     = (const float*)d_in[12];
    const float* W_ih   = (const float*)d_in[13];  // (1536,1285)
    const float* W_hh   = (const float*)d_in[14];  // (1536,512)
    const float* b_ih   = (const float*)d_in[15];
    const float* b_hh   = (const float*)d_in[16];
    const float* W_proj = (const float*)d_in[17];  // (1536,3000)
    const float* b_proj = (const float*)d_in[18];
    const float* Wc2    = (const float*)d_in[19];  // (512,512)
    const float* bc2    = (const float*)d_in[20];

    // workspace layout (floats), ~4.4 MB total
    float* ws = (float*)d_ws;
    float* memb   = ws;                  // 64*256
    float* hWz    = memb   + 16384;      // 64*512
    float* hWu    = hWz    + 32768;
    float* e_z    = hWu    + 32768;      // [t*64+b]
    float* e_u    = e_z    + 32768;
    float* zctx   = e_u    + 32768;      // 64*512
    float* uctx   = zctx   + 32768;
    float* cs     = uctx   + 32768;      // [t*64+b]
    float* ecs    = cs     + 32768;      // 64*512
    float* mbuf   = ecs    + 32768;      // 64 (pad 64)
    float* gruin  = mbuf   + 64;         // 64*1285
    float* gi     = gruin  + 64 * 1285;  // 64*1536
    float* gh     = gi     + 64 * 1536;
    float* h1     = gh     + 64 * 1536;  // 64*512
    float* projin = h1     + 32768;      // 64*1536
    float* gen    = projin + 64 * 1536;  // 64*3000
    float* zcopy  = gen    + 64 * 3000;  // 64*3512

    float* out_proba = (float*)d_out;                 // 64*3512
    float* out_h1a   = out_proba + 64 * VT;           // 64*512
    float* out_h1b   = out_h1a + 64 * 512;            // 64*512

    // 0) zero the atomic-reduction targets
    k_zero<<<(32768 + 255) / 256, 256, 0, stream>>>(e_z, 32768);
    k_zero<<<(32768 + 255) / 256, 256, 0, stream>>>(e_u, 32768);
    k_zero<<<(32768 + 255) / 256, 256, 0, stream>>>(cs, 32768);

    // 1) embedding gather
    k_embed<<<64, 256, 0, stream>>>(m_t, embW, memb);

    // 2) hWz = h0 @ Wz[:512] + bz ; hWu = h0 @ Wu[:512] + bu    (aligned, unguarded)
    gemm_wmma<0, false, false><<<dim3(8, 1), 128, 0, stream>>>(
        h0, Hh, Wz, Hh, hWz, Hh, bz, nullptr, 0, nullptr, 0, nullptr, 64, Hh, Hh, 64);
    gemm_wmma<0, false, false><<<dim3(8, 1), 128, 0, stream>>>(
        h0, Hh, Wu, Hh, hWu, Hh, bu, nullptr, 0, nullptr, 0, nullptr, 64, Hh, Hh, 64);

    // 3) e[t*64+b] = sum_n tanh((enc@W_bot)[r,n] + hW[b,n]) * v[n]   (fused, unguarded)
    gemm_wmma<1, false, false><<<dim3(8, 512), 128, 0, stream>>>(
        z_enc, Hh, Wz + (long)Hh * Hh, Hh, nullptr, 0, nullptr,
        hWz, Hh, vz, 0, e_z, TZ * Bz, Hh, Hh, 64);
    gemm_wmma<1, false, false><<<dim3(8, 512), 128, 0, stream>>>(
        u_enc, Hh, Wu + (long)Hh * Hh, Hh, nullptr, 0, nullptr,
        hWu, Hh, vu, 0, e_u, TU * Bz, Hh, Hh, 64);

    // 4) softmax over t + context
    k_attn_ctx<<<64, 256, 0, stream>>>(e_z, z_enc, zctx);
    k_attn_ctx<<<64, 256, 0, stream>>>(e_u, u_enc, uctx);

    // 5) GRU input assembly and gates
    k_concat_gru<<<64, 256, 0, stream>>>(memb, uctx, zctx, degree, gruin);
    gemm_wmma<0, true, true><<<dim3(24, 1), 128, 0, stream>>>(   // gi = x @ W_ih.T + b_ih (K=1285 ragged)
        gruin, Ee + 2 * Hh + Dd, W_ih, Ee + 2 * Hh + Dd, gi, 3 * Hh, b_ih,
        nullptr, 0, nullptr, 0, nullptr, 64, 3 * Hh, Ee + 2 * Hh + Dd, 64);
    gemm_wmma<0, true, false><<<dim3(24, 1), 128, 0, stream>>>(  // gh = h0 @ W_hh.T + b_hh (aligned)
        h0, Hh, W_hh, Hh, gh, 3 * Hh, b_hh,
        nullptr, 0, nullptr, 0, nullptr, 64, 3 * Hh, Hh, 64);
    k_gru_gate<<<64, 256, 0, stream>>>(gi, gh, h0, h1, out_h1a, out_h1b);

    // 6) cs[t*64+b] = sum_k tanh((z_enc@Wc2)[r,k] + bc2[k]) * h1[b,k]  (fused, unguarded)
    gemm_wmma<1, false, false><<<dim3(8, 512), 128, 0, stream>>>(
        z_enc, Hh, Wc2, Hh, nullptr, 0, nullptr,
        bc2, 0, h1, Hh, cs, TZ * Bz, Hh, Hh, 64);

    // 7) gen_score = [zctx,uctx,h1] @ W_proj + b_proj   (N=3000 ragged -> guarded)
    k_concat_proj<<<64, 256, 0, stream>>>(zctx, uctx, h1, projin);
    gemm_wmma<0, false, true><<<dim3(47, 1), 128, 0, stream>>>(
        projin, 3 * Hh, W_proj, Vv, gen, Vv, b_proj,
        nullptr, 0, nullptr, 0, nullptr, 64, Vv, 3 * Hh, 64);

    // 8) copy distribution: ecs, then HBM-streaming GEMV over sparse (460 MB)
    k_ecs<<<64, 256, 0, stream>>>(cs, ecs, mbuf);
    k_copy<<<dim3((VT / 4 + 255) / 256, 64), 256, 0, stream>>>(ecs, mbuf, sparse, zcopy);

    // 9) joint softmax over 6512 logits, combine gen+copy -> proba
    k_final<<<64, 256, 0, stream>>>(gen, zcopy, out_proba);
}
